// Chfloss_20478404067559
// MI455X (gfx1250) — matrix operand order repaired and checked by reading, not compiled, not run on hardware
//
#include <hip/hip_runtime.h>
#include <math.h>

typedef __attribute__((ext_vector_type(16))) _Float16 v16h;
typedef __attribute__((ext_vector_type(8)))  _Float16 v8h;
typedef __attribute__((ext_vector_type(4)))  _Float16 v4h;
typedef __attribute__((ext_vector_type(8)))  float    v8f;

#define CHF_STEPI 30
#define CHF_TIKF  0.1f
#define SAMPLE_STEPF 8.0f

// ---- WMMA fragment helpers (CDNA5 ISA 7.12.2 layouts, wave32) ----
// All LDS matrices are stored so that every fragment access is a contiguous,
// 16B-aligned ds_load_b128 / ds_store_b128.

// A operand (16x32 f16) from row-major M[m][k] (ld=64 halves):
// lane: m = lane&15, hi = lane>>4; halves j<8 -> K = k0+8*hi+j (contiguous),
// halves j>=8 -> K = k0+16+8*hi+(j-8) (contiguous).
__device__ __forceinline__ v16h load_a(const _Float16* m, int row0, int k0, int lane) {
  const int mm = lane & 15, hi = lane >> 4;
  const _Float16* p = m + (row0 + mm) * 64 + k0 + 8 * hi;
  const v8h lo  = *(const v8h*)p;          // 16B aligned
  const v8h hi8 = *(const v8h*)(p + 16);   // 16B aligned
  v16h a;
#pragma unroll
  for (int j = 0; j < 8; ++j) { a[j] = lo[j]; a[j + 8] = hi8[j]; }
  return a;
}

// B operand (32x16 f16) from column-of-B-major storage Bmat[n][k] (ld=64):
// lane: n = lane&15, hi = lane>>4; halves j<8 -> K = k0+8*hi+j, j>=8 -> K = k0+16+8*hi+(j-8).
__device__ __forceinline__ v16h load_b(const _Float16* bm, int col0, int k0, int lane) {
  const int n = lane & 15, hi = lane >> 4;
  const _Float16* p = bm + (col0 + n) * 64 + k0 + 8 * hi;
  const v8h lo  = *(const v8h*)p;          // 16B aligned
  const v8h hi8 = *(const v8h*)(p + 16);   // 16B aligned
  v16h b;
#pragma unroll
  for (int j = 0; j < 8; ++j) { b[j] = lo[j]; b[j + 8] = hi8[j]; }
  return b;
}

__device__ __forceinline__ v8f wmma16(v16h a, v16h b, v8f c) {
  // emits v_wmma_f32_16x16x32_f16
  return __builtin_amdgcn_wmma_f32_16x16x32_f16(
      /*neg_a=*/false, a, /*neg_b=*/false, b,
      /*c_mod=*/(short)0, c, /*reuse_a=*/false, /*reuse_b=*/false);
}

// One workgroup (256 threads = 8 wave32) per batch element.
__global__ __launch_bounds__(256)
void chf_loss_kernel(const float* __restrict__ dnn, const float* __restrict__ gt,
                     float* __restrict__ partial) {
  // x[w] == y[h] axis and shared r  =>  Cx == Cy, Sx == Sy: one table pair.
  __shared__ _Float16 Dh [64 * 64]; // D[h][w] = dnn - gt        (A-major)
  __shared__ _Float16 Cm [64 * 64]; // Cm[q][w] = cos(r[q]*x[w]) (B-major for stage1, A-major for stage2)
  __shared__ _Float16 Sm [64 * 64]; // sin(...)
  __shared__ _Float16 TcT[64 * 64]; // stage-1 out, transposed: TcT[q][h] (B-major for stage2)
  __shared__ _Float16 TsT[64 * 64];
  __shared__ float red[8];

  const int b   = blockIdx.x;
  const int tid = threadIdx.x;

  // Phase 0: difference image, vectorized (float4 in, 4 halves out).
  const float4* ob4 = (const float4*)(dnn + b * 4096);
  const float4* gb4 = (const float4*)(gt  + b * 4096);
#pragma unroll
  for (int i = tid; i < 1024; i += 256) {
    const float4 o = ob4[i], g = gb4[i];
    v4h d;
    d[0] = (_Float16)(o.x - g.x); d[1] = (_Float16)(o.y - g.y);
    d[2] = (_Float16)(o.z - g.z); d[3] = (_Float16)(o.w - g.w);
    *(v4h*)(Dh + i * 4) = d;
  }

  // Phase 1: one trig table pair; each thread fills 16 contiguous halves/table.
  {
    const int row = tid >> 2;          // frequency index p/q, 0..63 (>=60 zero-pad)
    const int c0  = (tid & 3) * 16;    // sample index base w/h
    v8h clo = {}, chi = {}, slo = {}, shi = {};
    if (row < 60) {
      const float r = (float)(row - CHF_STEPI) * CHF_TIKF;
#pragma unroll
      for (int j = 0; j < 8; ++j) {
        float s, c;
        const float x0 = SAMPLE_STEPF * (float)(c0 + j) + SAMPLE_STEPF * 0.5f;
        __sincosf(r * x0, &s, &c);
        clo[j] = (_Float16)c; slo[j] = (_Float16)s;
        const float x1 = SAMPLE_STEPF * (float)(c0 + 8 + j) + SAMPLE_STEPF * 0.5f;
        __sincosf(r * x1, &s, &c);
        chi[j] = (_Float16)c; shi[j] = (_Float16)s;
      }
    }
    _Float16* cp = Cm + row * 64 + c0;
    _Float16* sp = Sm + row * 64 + c0;
    *(v8h*)cp = clo; *(v8h*)(cp + 8) = chi;
    *(v8h*)sp = slo; *(v8h*)(sp + 8) = shi;
  }
  __syncthreads();

  const int wave = tid >> 5;
  const int lane = tid & 31;
  const int mm = lane & 15, hi = lane >> 4;

  // Phase 2 (stage-1): Tc = D x C^T, Ts = D x S^T (64x64x64 each); store transposed.
#pragma unroll
  for (int i = 0; i < 4; ++i) {
    const int job = wave + 8 * i;
    const _Float16* Bm = (job < 16) ? Cm  : Sm;
    _Float16*       Om = (job < 16) ? TcT : TsT;
    const int t = job & 15, tm = (t >> 2) * 16, tn = (t & 3) * 16; // tm: h-tile, tn: q-tile
    v8f c = {};
#pragma unroll
    for (int kb = 0; kb < 2; ++kb)
      c = wmma16(load_a(Dh, tm, kb * 32, lane), load_b(Bm, tn, kb * 32, lane), c);
    // C/D element j: (m = tm+j+8*hi, n = tn+mm). Store transposed -> contiguous b128.
    v8h h8;
#pragma unroll
    for (int j = 0; j < 8; ++j) h8[j] = (_Float16)c[j];
    *(v8h*)(Om + (tn + mm) * 64 + tm + 8 * hi) = h8;
  }
  __syncthreads();

  // Phase 3 (stage-2): R = C*Tc - S*Ts ; I = S*Tc + C*Ts ; sum of squares on the fly.
  float local = 0.f;
#pragma unroll
  for (int i = 0; i < 4; ++i) {
    const int job = wave + 8 * i;
    const bool isI = (job >= 16);
    const int t = job & 15, tm = (t >> 2) * 16, tn = (t & 3) * 16; // tm: p-tile, tn: q-tile
    v8f c = {};
#pragma unroll
    for (int kb = 0; kb < 2; ++kb)
      c = wmma16(load_a(isI ? Sm : Cm, tm, kb * 32, lane),
                 load_b(TcT, tn, kb * 32, lane), c);
#pragma unroll
    for (int kb = 0; kb < 2; ++kb) {
      // F16 WMMA NEG is C-negate only; negate the A fragment for the -S*Ts term.
      v16h a = load_a(isI ? Cm : Sm, tm, kb * 32, lane);
      if (!isI) a = -a;
      c = wmma16(a, load_b(TsT, tn, kb * 32, lane), c);
    }
#pragma unroll
    for (int j = 0; j < 8; ++j) local += c[j] * c[j];
  }

  // Phase 4: deterministic reduction (shuffle tree in-wave, serial across waves).
#pragma unroll
  for (int off = 16; off > 0; off >>= 1)
    local += __shfl_down(local, off, 32);
  if (lane == 0) red[wave] = local;
  __syncthreads();
  if (tid == 0) {
    float s = 0.f;
#pragma unroll
    for (int w = 0; w < 8; ++w) s += red[w];
    partial[b] = sqrtf(s) * CHF_TIKF;  // ||diff||_2 * CHF_TIK for this batch
  }
}

__global__ void chf_finalize(const float* __restrict__ partial,
                             float* __restrict__ outv, int B) {
  if (threadIdx.x == 0 && blockIdx.x == 0) {
    float s = 0.f;
    for (int i = 0; i < B; ++i) s += partial[i];
    outv[0] = s / (float)B;
  }
}

extern "C" void kernel_launch(void* const* d_in, const int* in_sizes, int n_in,
                              void* d_out, int out_size, void* d_ws, size_t ws_size,
                              hipStream_t stream) {
  (void)n_in; (void)out_size; (void)ws_size;
  const float* dnn = (const float*)d_in[0];
  const float* gt  = (const float*)d_in[1];
  float* out = (float*)d_out;
  float* ws  = (float*)d_ws;                 // B partial norms
  const int B = in_sizes[0] / 4096;          // 8 for the reference shapes

  chf_loss_kernel<<<B, 256, 0, stream>>>(dnn, gt, ws);
  chf_finalize<<<1, 32, 0, stream>>>(ws, out, B);
}
